// Block3D_47382079209488
// MI455X (gfx1250) — matrix operand order repaired and checked by reading, not compiled, not run on hardware
//
#include <hip/hip_runtime.h>
#include <math.h>

// ---------------- constants ----------------
constexpr int kT   = 16;
constexpr int kHW  = 4096;           // 64*64
constexpr int kN   = 65536;          // 16*64*64 positions
constexpr int kC   = 192;
constexpr int kC2  = 96;
constexpr int kC6  = 32;
constexpr int kHID = 768;
constexpr int XLD  = 200;            // xm LDS row stride (halves), 192 + pad
constexpr int HLD  = 776;            // h  LDS row stride (halves), 768 + pad
constexpr unsigned kStageLdsOff = 128u * XLD * 2u;   // byte offset of stage/hb region = 51200

typedef __attribute__((ext_vector_type(16))) __bf16 v16bf;
typedef __attribute__((ext_vector_type(8)))  __bf16 v8bf;
typedef __attribute__((ext_vector_type(8)))  float  v8f;
typedef __attribute__((ext_vector_type(4)))  unsigned int u32x4;
typedef __attribute__((ext_vector_type(8)))  unsigned int u32x8;

// ---------------- WMMA tile helpers (CDNA5 16x16x32 bf16 layouts, ISA 7.12.2) ----------------
__device__ inline v16bf load_a_tile(const __bf16* __restrict__ W, int ld, int m0, int k0, int lane) {
  int row = m0 + (lane & 15);
  int off = (lane & 16) ? 8 : 0;
  const __bf16* p = W + (size_t)row * ld + k0 + off;
  v8bf lo = *(const v8bf*)(p);
  v8bf hi = *(const v8bf*)(p + 16);
  return __builtin_shufflevector(lo, hi, 0,1,2,3,4,5,6,7,8,9,10,11,12,13,14,15);
}
__device__ inline v16bf load_b_tile(const __bf16* __restrict__ X, int ld, int n0, int k0, int lane) {
  int col = n0 + (lane & 15);
  int off = (lane & 16) ? 8 : 0;
  const __bf16* p = X + (size_t)col * ld + k0 + off;
  v8bf lo = *(const v8bf*)(p);
  v8bf hi = *(const v8bf*)(p + 16);
  return __builtin_shufflevector(lo, hi, 0,1,2,3,4,5,6,7,8,9,10,11,12,13,14,15);
}
__device__ inline v8f wmma_bf16(v16bf a, v16bf b, v8f c) {
  return __builtin_amdgcn_wmma_f32_16x16x32_bf16(false, a, false, b, (short)0, c, false, false);
}

// ---------------- f32 -> bf16 conversion ----------------
__global__ void cvt_bf16_kernel(const float* __restrict__ s, __bf16* __restrict__ d, int n) {
  for (int i = blockIdx.x * blockDim.x + threadIdx.x; i < n; i += gridDim.x * blockDim.x)
    d[i] = (__bf16)s[i];
}

// ---------------- LayerNorm1 over channels 96..191 -> xn bf16 [n][96] ----------------
__global__ void ln1_kernel(const float* __restrict__ x, const float* __restrict__ g,
                           const float* __restrict__ b, __bf16* __restrict__ xn) {
  int n = blockIdx.x * blockDim.x + threadIdx.x;
  const float* xp = x + (size_t)kC2 * kN + n;
  float s = 0.f, q = 0.f;
  for (int c = 0; c < kC2; ++c) { float v = xp[(size_t)c * kN]; s += v; q += v * v; }
  float mu = s * (1.f / kC2);
  float rs = rsqrtf(q * (1.f / kC2) - mu * mu + 1e-6f);
  __bf16* xo = xn + (size_t)n * kC2;
  for (int c = 0; c < kC2; ++c) {
    float v = xp[(size_t)c * kN];
    xo[c] = (__bf16)(g[c] * ((v - mu) * rs) + b[c]);
  }
}

// ---------------- branch conv stage: y_j = W_j(32x32) @ xn_i + b_j, j=0..2 ----------------
__global__ void conv_y_kernel(const __bf16* __restrict__ xn, int k0,
                              const __bf16* __restrict__ wb,          // 3 * 1024 bf16
                              const float* __restrict__ bias0, const float* __restrict__ bias1,
                              const float* __restrict__ bias2,
                              float* __restrict__ y) {                // [96][N]
  int lane = threadIdx.x & 31;
  int wave = threadIdx.x >> 5;
  int tile = blockIdx.x * (blockDim.x >> 5) + wave;
  int n0 = tile * 16;
  const float* bias[3] = {bias0, bias1, bias2};
  v16bf B = load_b_tile(xn, kC2, n0, k0, lane);
  int col  = n0 + (lane & 15);
  int rsub = (lane & 16) ? 8 : 0;
#pragma unroll
  for (int j = 0; j < 3; ++j) {
#pragma unroll
    for (int h = 0; h < 2; ++h) {
      v16bf A = load_a_tile(wb + j * 1024, 32, h * 16, 0, lane);
      v8f acc = {};
      acc = wmma_bf16(A, B, acc);
#pragma unroll
      for (int r = 0; r < 8; ++r) {
        int c = h * 16 + rsub + r;
        y[(size_t)(j * 32 + c) * kN + col] = acc[r] + bias[j][c];
      }
    }
  }
}

// ---------------- reductions (raw sums; scaling folded at product time) ----------------
__global__ void red_a1_kernel(const float* __restrict__ y1, float* __restrict__ a1) {
  __shared__ float s[256];
  int ct = blockIdx.x;                       // c*16 + t
  const float* p = y1 + (size_t)(ct >> 4) * kN + (ct & 15) * kHW;
  float acc = 0.f;
  for (int i = threadIdx.x; i < kHW; i += 256) acc += p[i];
  s[threadIdx.x] = acc; __syncthreads();
  for (int o = 128; o > 0; o >>= 1) { if (threadIdx.x < o) s[threadIdx.x] += s[threadIdx.x + o]; __syncthreads(); }
  if (threadIdx.x == 0) a1[ct] = s[0];
}
__global__ void red_b1_kernel(const float* __restrict__ y1, float* __restrict__ b1, int L) {
  int idx = blockIdx.x * blockDim.x + threadIdx.x;    // [S*32*4096)
  int hw = idx & 4095, c = (idx >> 12) & 31, b = idx >> 17;
  float s = 0.f;
  for (int t = 0; t < L; ++t) s += y1[(size_t)c * kN + (b * L + t) * kHW + hw];
  b1[idx] = s;
}
__global__ void red_a2_kernel(const float* __restrict__ y2, float* __restrict__ a2) {
  int idx = blockIdx.x * blockDim.x + threadIdx.x;    // 32768 = (c*16+t)*64+h
  int h = idx & 63, t = (idx >> 6) & 15, c = idx >> 10;
  const float* p = y2 + (size_t)c * kN + t * kHW + h * 64;
  float s = 0.f;
  for (int w = 0; w < 64; ++w) s += p[w];
  a2[idx] = s;
}
__global__ void red_a3_kernel(const float* __restrict__ y3, float* __restrict__ a3) {
  int idx = blockIdx.x * blockDim.x + threadIdx.x;    // (c*16+t)*64+w
  int w = idx & 63, t = (idx >> 6) & 15, c = idx >> 10;
  const float* p = y3 + (size_t)c * kN + t * kHW + w;
  float s = 0.f;
  for (int h = 0; h < 64; ++h) s += p[h * 64];
  a3[idx] = s;
}
__global__ void red_b2_kernel(const float* __restrict__ y2, float* __restrict__ b2, int L) {
  int idx = blockIdx.x * blockDim.x + threadIdx.x;    // b*2048 + c*64 + w
  int w = idx & 63, c = (idx >> 6) & 31, b = idx >> 11;
  float s = 0.f;
  for (int t = 0; t < L; ++t) {
    const float* p = y2 + (size_t)c * kN + (b * L + t) * kHW + w;
    for (int h = 0; h < 64; ++h) s += p[h * 64];
  }
  b2[idx] = s;
}
__global__ void red_b3_kernel(const float* __restrict__ y3, float* __restrict__ b3, int L) {
  int idx = blockIdx.x * blockDim.x + threadIdx.x;    // b*2048 + c*64 + h
  int h = idx & 63, c = (idx >> 6) & 31, b = idx >> 11;
  float s = 0.f;
  for (int t = 0; t < L; ++t) {
    const float* p = y3 + (size_t)c * kN + (b * L + t) * kHW + h * 64;
    for (int w = 0; w < 64; ++w) s += p[w];
  }
  b3[idx] = s;
}

// ---------------- out1*out2*out3 (separable outer products), pack bf16 [n][32] ----------------
__global__ void prod_kernel(const float* __restrict__ a1, const float* __restrict__ b1,
                            const float* __restrict__ a2, const float* __restrict__ b2,
                            const float* __restrict__ a3, const float* __restrict__ b3,
                            int L, float sA, float sB, __bf16* __restrict__ prod) {
  int idx = blockIdx.x * blockDim.x + threadIdx.x;    // n*32 + c
  int c = idx & 31, n = idx >> 5;
  int t = n >> 12, hw = n & 4095, h = hw >> 6, w = hw & 63, b = t / L;
  float v1 = a1[c * 16 + t] * b1[((b * 32 + c) << 12) + hw] * sA;
  float v2 = a2[(c * 16 + t) * 64 + h] * b2[(b * 32 + c) * 64 + w] * sB;
  float v3 = a3[(c * 16 + t) * 64 + w] * b3[(b * 32 + c) * 64 + h] * sB;
  prod[(size_t)n * 32 + c] = (__bf16)(v1 * v2 * v3);
}

// ---------------- wp GEMM + shortcut -> out1 channels [ch0, ch0+32) ----------------
__global__ void wp_kernel(const __bf16* __restrict__ prod, const __bf16* __restrict__ wp,
                          const float* __restrict__ bp, const float* __restrict__ x,
                          float* __restrict__ out1, int ch0) {
  int lane = threadIdx.x & 31;
  int wave = threadIdx.x >> 5;
  int tile = blockIdx.x * (blockDim.x >> 5) + wave;
  int n0 = tile * 16;
  v16bf B = load_b_tile(prod, 32, n0, 0, lane);
  int col  = n0 + (lane & 15);
  int rsub = (lane & 16) ? 8 : 0;
#pragma unroll
  for (int h = 0; h < 2; ++h) {
    v16bf A = load_a_tile(wp, 32, h * 16, 0, lane);
    v8f acc = {};
    acc = wmma_bf16(A, B, acc);
#pragma unroll
    for (int r = 0; r < 8; ++r) {
      int c = h * 16 + rsub + r;
      size_t o = (size_t)(ch0 + c) * kN + col;
      out1[o] = x[o] + acc[r] + bp[c];
    }
  }
}

// ---------------- fused LN2 + MLP (192->768 GELU ->192) + residual ----------------
// TDM stages the 192x128 f32 out1 tile into LDS (one tensor_load_to_lds per WG),
// LN2 consumes it from LDS, then the region is reused as the bf16 hidden buffer.
__global__ __launch_bounds__(256)
void mlp_kernel(const float* __restrict__ out1,
                const float* __restrict__ g, const float* __restrict__ be,
                const __bf16* __restrict__ fc1w, const float* __restrict__ fc1b,
                const __bf16* __restrict__ fc2w, const float* __restrict__ fc2b,
                float* __restrict__ out) {
  extern __shared__ __align__(16) char smem_raw[];
  __bf16* xm    = (__bf16*)smem_raw;             // [128][XLD]
  __bf16* hb    = xm + 128 * XLD;                // [128][HLD]  (byte off kStageLdsOff)
  float*  stage = (float*)hb;                    // TDM-staged out1 tile: [192][128] f32
  float*  red   = (float*)(hb + 128 * HLD);      // [4][128]

  const int tid = threadIdx.x;
  const int n0  = blockIdx.x * 128;
  const int lane = tid & 31;
  const int wv   = tid >> 5;

  // ---- TDM: stage out1[0:192][n0:n0+128] (f32) into LDS, one wave issues ----
  if (wv == 0) {
    unsigned long long ga = (unsigned long long)(out1 + n0);
    u32x4 g0;
    g0.x = 1u;                                   // count=1, user descriptor
    g0.y = kStageLdsOff;                         // lds_addr (bytes)
    g0.z = (unsigned)(ga & 0xffffffffu);         // global_addr[31:0]
    g0.w = ((unsigned)(ga >> 32) & 0x01ffffffu) | (2u << 30);  // addr[56:32] | type=2
    u32x8 g1;
    g1.s0 = 2u << 16;                            // data_size=4B; no mask/pad/iterate
    g1.s1 = 0u;                                  // barrier addr 0 | tensor_dim0[15:0]=0 (65536)
    g1.s2 = 1u | (192u << 16);                   // tensor_dim0[31:16]=1 | tensor_dim1[15:0]=192
    g1.s3 = 128u << 16;                          // tensor_dim1[31:16]=0 | tile_dim0=128
    g1.s4 = 192u;                                // tile_dim1=192 | tile_dim2=0
    g1.s5 = 65536u;                              // tensor_dim0_stride[31:0]
    g1.s6 = 0u;                                  // stride[47:32]=0 | dim1_stride[15:0]=0
    g1.s7 = 0u;
    asm volatile("tensor_load_to_lds %0, %1" :: "s"(g0), "s"(g1) : "memory");
    __builtin_amdgcn_s_wait_tensorcnt(0);
  }
  __syncthreads();

  // ---- LN2 over 192 channels from the staged LDS tile, 2 threads per position ----
  {
    int p = tid & 127, half = tid >> 7;
    float s = 0.f, q = 0.f;
    for (int c = half * 96; c < half * 96 + 96; ++c) {
      float v = stage[c * 128 + p]; s += v; q += v * v;
    }
    red[half * 128 + p] = s;
    red[256 + half * 128 + p] = q;
    __syncthreads();
    float st = red[p] + red[128 + p];
    float qt = red[256 + p] + red[384 + p];
    float mu = st * (1.f / kC);
    float rs = rsqrtf(qt * (1.f / kC) - mu * mu + 1e-6f);
    for (int c = half * 96; c < half * 96 + 96; ++c) {
      float v = stage[c * 128 + p];
      xm[p * XLD + c] = (__bf16)(g[c] * ((v - mu) * rs) + be[c]);
    }
  }
  __syncthreads();    // stage region now dead; reused as hb below

  const int col  = lane & 15;
  const int rsub = (lane & 16) ? 8 : 0;

  // ---- GEMM1: h(768 x 128) = fc1(768x192) @ xm ; each wave owns 6 of 48 M-tiles ----
  for (int item = wv; item < 48; item += 8) {
    int m0 = item * 16;
    v8f acc[8] = {};
    for (int k = 0; k < 6; ++k) {
      v16bf A = load_a_tile(fc1w, kC, m0, k * 32, lane);
#pragma unroll
      for (int n = 0; n < 8; ++n) {
        v16bf B = load_b_tile(xm, XLD, n * 16, k * 32, lane);
        acc[n] = wmma_bf16(A, B, acc[n]);
      }
    }
#pragma unroll
    for (int n = 0; n < 8; ++n)
#pragma unroll
      for (int r = 0; r < 8; ++r) {
        int row = m0 + rsub + r;
        float v = acc[n][r] + fc1b[row];
        v = 0.5f * v * (1.f + erff(v * 0.70710678118654752f));   // exact GELU
        hb[(n * 16 + col) * HLD + row] = (__bf16)v;
      }
  }
  __syncthreads();

  // ---- GEMM2: out(192 x 128) = fc2(192x768) @ h ; 24 (M x N-half) items, 3 per wave ----
  for (int item = wv; item < 24; item += 8) {
    int m0 = (item >> 1) * 16;
    int nh = (item & 1) * 4;
    v8f acc[4] = {};
    for (int k = 0; k < 24; ++k) {
      v16bf A = load_a_tile(fc2w, kHID, m0, k * 32, lane);
#pragma unroll
      for (int n = 0; n < 4; ++n) {
        v16bf B = load_b_tile(hb, HLD, (nh + n) * 16, k * 32, lane);
        acc[n] = wmma_bf16(A, B, acc[n]);
      }
    }
#pragma unroll
    for (int n = 0; n < 4; ++n)
#pragma unroll
      for (int r = 0; r < 8; ++r) {
        int row = m0 + rsub + r;
        size_t o = (size_t)row * kN + n0 + (nh + n) * 16 + col;
        out[o] = acc[n][r] + fc2b[row] + out1[o];
      }
  }
}

// ---------------- host launcher ----------------
extern "C" void kernel_launch(void* const* d_in, const int* in_sizes, int n_in,
                              void* d_out, int out_size, void* d_ws, size_t ws_size,
                              hipStream_t stream) {
  (void)in_sizes; (void)n_in; (void)out_size; (void)ws_size;
  const float* x   = (const float*)d_in[0];
  const float* n1w = (const float*)d_in[1];
  const float* n1b = (const float*)d_in[2];
  const float* n2w = (const float*)d_in[3];
  const float* n2b = (const float*)d_in[4];
  // ma blocks: base 5 + 8*i : w1,b1,w2,b2,w3,b3,wp,bp
  const float* fc1w = (const float*)d_in[29];
  const float* fc1b = (const float*)d_in[30];
  const float* fc2w = (const float*)d_in[31];
  const float* fc2b = (const float*)d_in[32];
  float* out = (float*)d_out;

  // workspace bump allocator
  char* wsp = (char*)d_ws;
  auto alloc = [&](size_t bytes) -> void* {
    void* p = (void*)wsp;
    wsp += (bytes + 255) & ~(size_t)255;
    return p;
  };
  __bf16* xn    = (__bf16*)alloc((size_t)kN * kC2 * 2);
  __bf16* wbf   = (__bf16*)alloc(12 * 1024 * 2);           // [branch][w1,w2,w3,wp]
  __bf16* fc1bf = (__bf16*)alloc((size_t)kHID * kC * 2);
  __bf16* fc2bf = (__bf16*)alloc((size_t)kC * kHID * 2);
  float*  ybuf  = (float*)alloc((size_t)3 * 32 * kN * 4);  // y1|y2|y3 for one branch
  float*  a1    = (float*)alloc(512 * 4);
  float*  b1    = (float*)alloc((size_t)4 * 32 * 4096 * 4);
  float*  a2    = (float*)alloc(32768 * 4);
  float*  b2    = (float*)alloc(8192 * 4);
  float*  a3    = (float*)alloc(32768 * 4);
  float*  b3    = (float*)alloc(8192 * 4);
  __bf16* prod  = (__bf16*)alloc((size_t)kN * 32 * 2);
  float*  out1  = (float*)alloc((size_t)kC * kN * 4);

  auto cvt = [&](const float* s, __bf16* d, int n) {
    cvt_bf16_kernel<<<dim3((n + 255) / 256), dim3(256), 0, stream>>>(s, d, n);
  };
  for (int i = 0; i < 3; ++i)
    for (int j = 0; j < 4; ++j)
      cvt((const float*)d_in[5 + 8 * i + 2 * j], wbf + i * 4096 + j * 1024, 1024);
  cvt(fc1w, fc1bf, kHID * kC);
  cvt(fc2w, fc2bf, kC * kHID);

  ln1_kernel<<<dim3(kN / 256), dim3(256), 0, stream>>>(x, n1w, n1b, xn);

  // out1[0:96] = x_no
  hipMemcpyAsync(out1, x, (size_t)kC2 * kN * 4, hipMemcpyDeviceToDevice, stream);

  const int Ls[3] = {4, 8, 16};
  for (int i = 0; i < 3; ++i) {
    const int L = Ls[i], S = 16 / L;
    const float* mb1 = (const float*)d_in[5 + 8 * i + 1];
    const float* mb2 = (const float*)d_in[5 + 8 * i + 3];
    const float* mb3 = (const float*)d_in[5 + 8 * i + 5];
    const float* mbp = (const float*)d_in[5 + 8 * i + 7];

    conv_y_kernel<<<dim3(1024), dim3(128), 0, stream>>>(xn, i * 32, wbf + i * 4096,
                                                        mb1, mb2, mb3, ybuf);
    red_a1_kernel<<<dim3(512), dim3(256), 0, stream>>>(ybuf, a1);
    red_b1_kernel<<<dim3(S * 32 * 4096 / 256), dim3(256), 0, stream>>>(ybuf, b1, L);
    red_a2_kernel<<<dim3(128), dim3(256), 0, stream>>>(ybuf + (size_t)32 * kN, a2);
    red_b2_kernel<<<dim3(S * 8), dim3(256), 0, stream>>>(ybuf + (size_t)32 * kN, b2, L);
    red_a3_kernel<<<dim3(128), dim3(256), 0, stream>>>(ybuf + (size_t)64 * kN, a3);
    red_b3_kernel<<<dim3(S * 8), dim3(256), 0, stream>>>(ybuf + (size_t)64 * kN, b3, L);

    float sA = 1.f / (4096.f * (float)L);
    float sB = 1.f / (64.f * (float)L * 64.f);
    prod_kernel<<<dim3(kN * 32 / 256), dim3(256), 0, stream>>>(a1, b1, a2, b2, a3, b3,
                                                               L, sA, sB, prod);
    wp_kernel<<<dim3(1024), dim3(128), 0, stream>>>(prod, wbf + i * 4096 + 3072, mbp,
                                                    x, out1, kC2 + i * 32);
  }

  size_t smem = (size_t)128 * XLD * 2 + (size_t)128 * HLD * 2 + 512 * 4;  // ~246 KB < 320 KB
  mlp_kernel<<<dim3(kN / 128), dim3(256), smem, stream>>>(out1, n2w, n2b,
                                                          fc1bf, fc1b, fc2bf, fc2b, out);
}
